// TSPVectorGraphCombined_85057532330020
// MI455X (gfx1250) — matrix operand order repaired and checked by reading, not compiled, not run on hardware
//
#include <hip/hip_runtime.h>
#include <math.h>

// ---------------- constants (match reference) ----------------
#define NN   20000
#define EE   640000
#define UU   32
#define DD   12
#define PP   128
#define BNEPS 1e-5f

typedef __attribute__((ext_vector_type(16))) _Float16 v16h;
typedef __attribute__((ext_vector_type(8)))  float    v8f;

__device__ __forceinline__ float siluf(float x) { return x / (1.f + __expf(-x)); }
__device__ __forceinline__ float sigmf(float x) { return 1.f / (1.f + __expf(-x)); }

// ---------------- utility ----------------
__global__ void k_zero(float* __restrict__ p, int n) {
    int i = blockIdx.x * blockDim.x + threadIdx.x;
    if (i < n) p[i] = 0.f;
}

__global__ void k_degree(const int* __restrict__ src, float* __restrict__ deg, int E) {
    int e = blockIdx.x * blockDim.x + threadIdx.x;
    if (e < E) atomicAdd(&deg[src[e]], 1.0f);
}

// h[n,u] = silu(x[n,0]*w[u,0] + x[n,1]*w[u,1] + b[u])
__global__ void k_node_embed(const float* __restrict__ x, const float* __restrict__ w,
                             const float* __restrict__ b, float* __restrict__ h, int N) {
    int idx = blockIdx.x * blockDim.x + threadIdx.x;
    if (idx >= N * UU) return;
    int n = idx >> 5, u = idx & 31;
    float v = x[n * 2] * w[u * 2] + x[n * 2 + 1] * w[u * 2 + 1] + b[u];
    h[idx] = siluf(v);
}

// w[e,u] = silu(ea[e]*w0[u] + b[u])
__global__ void k_edge_embed(const float* __restrict__ ea, const float* __restrict__ w0,
                             const float* __restrict__ b, float* __restrict__ w, int E) {
    int idx = blockIdx.x * blockDim.x + threadIdx.x;
    if (idx >= E * UU) return;
    int e = idx >> 5, u = idx & 31;
    float v = ea[e] * w0[u] + b[u];
    w[idx] = siluf(v);
}

// ---------------- WMMA tile helpers ----------------
// A tile (16x32 f16) per ISA layout: lane<16 -> row=lane&15, K in {0..7,16..23};
// lane>=16 -> same row, K in {8..15,24..31}.
// B tile (32x16 f16): col = lane&15; lanes 0-15 hold K in {0..7,16..23}, lanes 16-31 K in {8..15,24..31}.
// D (f32): VGPR r -> (lanes 0-15: M=r, N=lane) (lanes 16-31: M=r+8, N=lane-16).

__device__ __forceinline__ void load_b_tiles(const float* __restrict__ W, int lane,
                                             v16h& b0, v16h& b1) {
    int col = lane & 15;
    int kbb = (lane >> 4) << 3;  // 0 or 8
#pragma unroll
    for (int j = 0; j < 16; ++j) {
        int kk = (j < 8) ? (kbb + j) : (kbb + 8 + j);
        b0[j] = (_Float16)W[col * UU + kk];          // output cols 0..15
        b1[j] = (_Float16)W[(col + 16) * UU + kk];   // output cols 16..31
    }
}

__device__ __forceinline__ void store_d(float* __restrict__ out, const float* __restrict__ bias,
                                        int r0, int lane, const v8f& c0, const v8f& c1) {
    int col = lane & 15;
    int mbase = r0 + ((lane >> 4) << 3);
    float bn0 = bias[col], bn1 = bias[col + 16];
#pragma unroll
    for (int r = 0; r < 8; ++r) {
        long orow = mbase + r;
        out[orow * UU + col]      = c0[r] + bn0;
        out[orow * UU + col + 16] = c1[r] + bn1;
    }
}

// ---------------- fused node GEMM: x{1..4} = h @ vW[i,k]^T + vB[i,k] ----------------
// One wave per 16-row tile; A loaded once, 8 WMMAs.
__global__ void k_node_gemm4(const float* __restrict__ h, const float* __restrict__ Wl,  // [4,32,32]
                             const float* __restrict__ Bl,                               // [4,32]
                             float* __restrict__ x1, float* __restrict__ x2,
                             float* __restrict__ x3, float* __restrict__ x4, int rows) {
    int wave = (int)((blockIdx.x * blockDim.x + threadIdx.x) >> 5);
    int lane = threadIdx.x & 31;
    int r0 = wave * 16;
    if (r0 >= rows) return;

    int arow = r0 + (lane & 15);
    int kb   = (lane >> 4) << 3;  // 0 or 8
    const float* ap = h + (long)arow * UU;
    v16h a;
#pragma unroll
    for (int j = 0; j < 8; ++j) {
        a[j]     = (_Float16)ap[kb + j];
        a[j + 8] = (_Float16)ap[kb + 16 + j];
    }

    float* outs[4] = {x1, x2, x3, x4};
#pragma unroll
    for (int k = 0; k < 4; ++k) {
        v16h b0, b1;
        load_b_tiles(Wl + (size_t)k * UU * UU, lane, b0, b1);
        v8f c0 = {}, c1 = {};
        c0 = __builtin_amdgcn_wmma_f32_16x16x32_f16(false, a, false, b0, (short)0, c0, false, false);
        c1 = __builtin_amdgcn_wmma_f32_16x16x32_f16(false, a, false, b1, (short)0, c1, false, false);
        store_d(outs[k], Bl + k * UU, r0, lane, c0, c1);
    }
}

// ---------------- fused edge GEMM + message scatter ----------------
// w1 = w @ eW^T + eB   AND   agg[src[e],u] += sigmoid(w[e,u]) * x2[dst[e],u]
// (the A-tile of the GEMM *is* w[e,u], so the message pass costs no extra read of w)
__global__ void k_edge_gemm_msg(const float* __restrict__ wfeat, const float* __restrict__ W,
                                const float* __restrict__ bias, float* __restrict__ w1,
                                const int* __restrict__ src, const int* __restrict__ dst,
                                const float* __restrict__ x2, float* __restrict__ agg,
                                int rows) {
    int wave = (int)((blockIdx.x * blockDim.x + threadIdx.x) >> 5);
    int lane = threadIdx.x & 31;
    int r0 = wave * 16;
    if (r0 >= rows) return;

    int arow = r0 + (lane & 15);
    int kb   = (lane >> 4) << 3;  // 0 or 8
    const float* ap = wfeat + (long)arow * UU;
    float av[16];
    v16h a;
#pragma unroll
    for (int j = 0; j < 8; ++j) {
        float v0 = ap[kb + j];
        float v1 = ap[kb + 16 + j];
        av[j] = v0; av[j + 8] = v1;
        a[j] = (_Float16)v0; a[j + 8] = (_Float16)v1;
    }

    v16h b0, b1;
    load_b_tiles(W, lane, b0, b1);
    v8f c0 = {}, c1 = {};
    c0 = __builtin_amdgcn_wmma_f32_16x16x32_f16(false, a, false, b0, (short)0, c0, false, false);
    c1 = __builtin_amdgcn_wmma_f32_16x16x32_f16(false, a, false, b1, (short)0, c1, false, false);
    store_d(w1, bias, r0, lane, c0, c1);

    // message + scatter using the already-loaded A values
    int sbase = src[arow] * UU;
    int dbase = dst[arow] * UU;
#pragma unroll
    for (int j = 0; j < 16; ++j) {
        int u = (j < 8) ? (kb + j) : (kb + 8 + j);  // j>=8 -> kb+16+(j-8)
        float m = sigmf(av[j]) * x2[dbase + u];
        atomicAdd(&agg[sbase + u], m);
    }
}

// ---------------- fused pre-BN + column stats ----------------
// tn = x1 + agg/max(deg,1); accumulate per-channel sum/sumsq (channel = idx&31)
__global__ void k_node_prebn(const float* __restrict__ x1, const float* __restrict__ agg,
                             const float* __restrict__ deg, float* __restrict__ tn,
                             float* __restrict__ sum, float* __restrict__ sumsq, int N) {
    int lane = threadIdx.x & 31;
    float s = 0.f, ss = 0.f;
    int total = N * UU, stride = gridDim.x * blockDim.x;
    for (int idx = blockIdx.x * blockDim.x + threadIdx.x; idx < total; idx += stride) {
        int n = idx >> 5;
        float d = deg[n]; d = d < 1.f ? 1.f : d;
        float v = x1[idx] + agg[idx] / d;
        tn[idx] = v;
        s += v; ss += v * v;
    }
    __shared__ float ls[256], lq[256];
    ls[threadIdx.x] = s; lq[threadIdx.x] = ss;
    __syncthreads();
    if (threadIdx.x < 32) {
        float a = 0.f, b = 0.f;
        for (int g = 0; g < 8; ++g) { a += ls[g * 32 + lane]; b += lq[g * 32 + lane]; }
        atomicAdd(&sum[lane], a);
        atomicAdd(&sumsq[lane], b);
    }
}

// te = w1 + x3[src] + x4[dst]  (in place over w1), with fused stats
__global__ void k_edge_prebn(float* __restrict__ w1, const float* __restrict__ x3,
                             const float* __restrict__ x4, const int* __restrict__ src,
                             const int* __restrict__ dst, float* __restrict__ sum,
                             float* __restrict__ sumsq, int E) {
    int lane = threadIdx.x & 31;
    float s = 0.f, ss = 0.f;
    int total = E * UU, stride = gridDim.x * blockDim.x;
    for (int idx = blockIdx.x * blockDim.x + threadIdx.x; idx < total; idx += stride) {
        int e = idx >> 5, u = idx & 31;
        float v = w1[idx] + x3[src[e] * UU + u] + x4[dst[e] * UU + u];
        w1[idx] = v;
        s += v; ss += v * v;
    }
    __shared__ float ls[256], lq[256];
    ls[threadIdx.x] = s; lq[threadIdx.x] = ss;
    __syncthreads();
    if (threadIdx.x < 32) {
        float a = 0.f, b = 0.f;
        for (int g = 0; g < 8; ++g) { a += ls[g * 32 + lane]; b += lq[g * 32 + lane]; }
        atomicAdd(&sum[lane], a);
        atomicAdd(&sumsq[lane], b);
    }
}

// dst += silu( (t - mean) * g * rsqrt(var+eps) + b )
__global__ void k_bn_silu_residual(float* __restrict__ dsta, const float* __restrict__ t,
                                   const float* __restrict__ sum, const float* __restrict__ sumsq,
                                   const float* __restrict__ g, const float* __restrict__ b,
                                   int rows) {
    int idx = blockIdx.x * blockDim.x + threadIdx.x;
    if (idx >= rows * UU) return;
    int u = idx & 31;
    float inv = 1.f / (float)rows;
    float m = sum[u] * inv;
    float var = sumsq[u] * inv - m * m;
    float v = (t[idx] - m) * (g[u] * rsqrtf(var + BNEPS)) + b[u];
    dsta[idx] += siluf(v);
}

// per-channel column sum (for h.mean(0))
__global__ void k_colsum(const float* __restrict__ x, int rows, float* __restrict__ sum) {
    int lane = threadIdx.x & 31;
    float s = 0.f;
    int total = rows * UU, stride = gridDim.x * blockDim.x;
    for (int idx = blockIdx.x * blockDim.x + threadIdx.x; idx < total; idx += stride)
        s += x[idx];
    __shared__ float ls[256];
    ls[threadIdx.x] = s;
    __syncthreads();
    if (threadIdx.x < 32) {
        float a = 0.f;
        for (int g = 0; g < 8; ++g) a += ls[g * 32 + lane];
        atomicAdd(&sum[lane], a);
    }
}

// ---------------- CNN branch ----------------
// out[co,l] = bias[co] + sum_ci sum_t in[ci, l+t-1] * w[co,ci,t]   (pad 1)
__global__ void k_conv1d3(const float* __restrict__ in, const float* __restrict__ wt,
                          const float* __restrict__ bias, float* __restrict__ out,
                          int Cin, int L, int Cout) {
    int idx = blockIdx.x * blockDim.x + threadIdx.x;
    if (idx >= Cout * L) return;
    int l = idx % L, co = idx / L;
    float acc = bias[co];
    const float* wp = wt + (long)co * Cin * 3;
    for (int ci = 0; ci < Cin; ++ci) {
        const float* ip = in + (long)ci * L + l;
        float xm = (l > 0)     ? ip[-1] : 0.f;
        float x0 = ip[0];
        float xp = (l < L - 1) ? ip[1]  : 0.f;
        acc += wp[ci * 3 + 0] * xm + wp[ci * 3 + 1] * x0 + wp[ci * 3 + 2] * xp;
    }
    out[idx] = acc;
}

// per-row (channel) mean/var over L; one block per channel
__global__ void k_stats_rows(const float* __restrict__ x, int L,
                             float* __restrict__ mean, float* __restrict__ var) {
    int c = blockIdx.x;
    float s = 0.f, ss = 0.f;
    for (int l = threadIdx.x; l < L; l += blockDim.x) {
        float v = x[(long)c * L + l];
        s += v; ss += v * v;
    }
    __shared__ float ls[256], lq[256];
    ls[threadIdx.x] = s; lq[threadIdx.x] = ss;
    __syncthreads();
    for (int off = 128; off > 0; off >>= 1) {
        if (threadIdx.x < off) { ls[threadIdx.x] += ls[threadIdx.x + off]; lq[threadIdx.x] += lq[threadIdx.x + off]; }
        __syncthreads();
    }
    if (threadIdx.x == 0) {
        float m = ls[0] / (float)L;
        mean[c] = m;
        var[c]  = lq[0] / (float)L - m * m;
    }
}

// out[c, l2] = 0.5*(silu(bn(y[c,2*l2])) + silu(bn(y[c,2*l2+1])))
__global__ void k_bn_silu_pool(const float* __restrict__ y, const float* __restrict__ mean,
                               const float* __restrict__ var, const float* __restrict__ g,
                               const float* __restrict__ b, float* __restrict__ out,
                               int C, int L) {
    int L2 = L >> 1;
    int idx = blockIdx.x * blockDim.x + threadIdx.x;
    if (idx >= C * L2) return;
    int l2 = idx % L2, c = idx / L2;
    float m = mean[c], sc = g[c] * rsqrtf(var[c] + BNEPS), bb = b[c];
    float a0 = (y[(long)c * L + 2 * l2]     - m) * sc + bb;
    float a1 = (y[(long)c * L + 2 * l2 + 1] - m) * sc + bb;
    out[idx] = 0.5f * (siluf(a0) + siluf(a1));
}

// omean[c] = mean_l silu(bn(y[c,l])) ; one block per channel
__global__ void k_bn_silu_mean(const float* __restrict__ y, const float* __restrict__ mean,
                               const float* __restrict__ var, const float* __restrict__ g,
                               const float* __restrict__ b, float* __restrict__ omean, int L) {
    int c = blockIdx.x;
    float m = mean[c], sc = g[c] * rsqrtf(var[c] + BNEPS), bb = b[c];
    float s = 0.f;
    for (int l = threadIdx.x; l < L; l += blockDim.x) {
        float v = (y[(long)c * L + l] - m) * sc + bb;
        s += siluf(v);
    }
    __shared__ float ls[256];
    ls[threadIdx.x] = s;
    __syncthreads();
    for (int off = 128; off > 0; off >>= 1) {
        if (threadIdx.x < off) ls[threadIdx.x] += ls[threadIdx.x + off];
        __syncthreads();
    }
    if (threadIdx.x == 0) omean[c] = ls[0] / (float)L;
}

// ---------------- combined head: one block of 128 threads (one per particle) ----------------
__global__ void k_head(const float* __restrict__ o3mean, const float* __restrict__ fc_w,
                       const float* __restrict__ fc_b, const float* __restrict__ hsum,
                       const float* __restrict__ lin1_w, const float* __restrict__ lin1_b,
                       const float* __restrict__ cbn_g, const float* __restrict__ cbn_b,
                       const float* __restrict__ lin2_w, const float* __restrict__ lin2_b,
                       float* __restrict__ out) {
    __shared__ float c1[PP][UU];
    __shared__ float mu[UU], scale[UU];
    int p = threadIdx.x;  // 0..127
    float c[UU];
    float o = o3mean[p];
    const float invN = 1.f / (float)NN;
#pragma unroll
    for (int u = 0; u < UU; ++u)
        c[u] = o * fc_w[u] + fc_b[u] + hsum[u] * invN;   // pop_embed + prob_embed
#pragma unroll
    for (int u2 = 0; u2 < UU; ++u2) {
        float acc = lin1_b[u2];
        for (int u = 0; u < UU; ++u) acc += c[u] * lin1_w[u2 * UU + u];
        c1[p][u2] = acc;
    }
    __syncthreads();
    if (p < UU) {
        float s = 0.f, ss = 0.f;
        for (int q = 0; q < PP; ++q) { float v = c1[q][p]; s += v; ss += v * v; }
        float m = s / (float)PP;
        float var = ss / (float)PP - m * m;
        mu[p] = m;
        scale[p] = cbn_g[p] * rsqrtf(var + BNEPS);
    }
    __syncthreads();
    float act[UU];
#pragma unroll
    for (int u = 0; u < UU; ++u)
        act[u] = siluf((c1[p][u] - mu[u]) * scale[u] + cbn_b[u]);
    float z[3];
    for (int j = 0; j < 3; ++j) {
        float acc = lin2_b[j];
        for (int u = 0; u < UU; ++u) acc += act[u] * lin2_w[j * UU + u];
        z[j] = acc;
    }
    float mx = fmaxf(z[0], fmaxf(z[1], z[2]));
    float e0 = __expf(z[0] - mx), e1 = __expf(z[1] - mx), e2 = __expf(z[2] - mx);
    float inv = 1.f / (e0 + e1 + e2);
    out[p * 3 + 0] = e0 * inv;
    out[p * 3 + 1] = e1 * inv;
    out[p * 3 + 2] = e2 * inv;
}

// ---------------- host launcher ----------------
static inline int ceil_div(int a, int b) { return (a + b - 1) / b; }

extern "C" void kernel_launch(void* const* d_in, const int* in_sizes, int n_in,
                              void* d_out, int out_size, void* d_ws, size_t ws_size,
                              hipStream_t stream) {
    (void)in_sizes; (void)n_in; (void)out_size; (void)ws_size;

    const float* x         = (const float*)d_in[0];
    const int*   ei        = (const int*)  d_in[1];
    const float* edge_attr = (const float*)d_in[2];
    const float* pop       = (const float*)d_in[3];
    const float* v_lin0_w  = (const float*)d_in[4];
    const float* v_lin0_b  = (const float*)d_in[5];
    const float* vW        = (const float*)d_in[6];   // [D,4,U,U]
    const float* vB        = (const float*)d_in[7];   // [D,4,U]
    const float* v_bn_g    = (const float*)d_in[8];
    const float* v_bn_b    = (const float*)d_in[9];
    const float* e_lin0_w  = (const float*)d_in[10];
    const float* e_lin0_b  = (const float*)d_in[11];
    const float* eW        = (const float*)d_in[12];  // [D,U,U]
    const float* eB        = (const float*)d_in[13];
    const float* e_bn_g    = (const float*)d_in[14];
    const float* e_bn_b    = (const float*)d_in[15];
    const float* conv1_w   = (const float*)d_in[16];
    const float* conv1_b   = (const float*)d_in[17];
    const float* bn1_g     = (const float*)d_in[18];
    const float* bn1_b     = (const float*)d_in[19];
    const float* conv2_w   = (const float*)d_in[20];
    const float* conv2_b   = (const float*)d_in[21];
    const float* bn2_g     = (const float*)d_in[22];
    const float* bn2_b     = (const float*)d_in[23];
    const float* conv3_w   = (const float*)d_in[24];
    const float* conv3_b   = (const float*)d_in[25];
    const float* bn3_g     = (const float*)d_in[26];
    const float* bn3_b     = (const float*)d_in[27];
    const float* fc_w      = (const float*)d_in[28];
    const float* fc_b      = (const float*)d_in[29];
    const float* lin1_w    = (const float*)d_in[30];
    const float* lin1_b    = (const float*)d_in[31];
    const float* cbn_g     = (const float*)d_in[32];
    const float* cbn_b     = (const float*)d_in[33];
    const float* lin2_w    = (const float*)d_in[34];
    const float* lin2_b    = (const float*)d_in[35];

    const int* src = ei;
    const int* dst = ei + EE;

    // workspace carve-up (floats)
    float* ws = (float*)d_ws;
    size_t off = 0;
    float* deg  = ws + off; off += NN;
    float* h    = ws + off; off += (size_t)NN * UU;
    float* x1   = ws + off; off += (size_t)NN * UU;
    float* x2   = ws + off; off += (size_t)NN * UU;
    float* x3   = ws + off; off += (size_t)NN * UU;
    float* x4   = ws + off; off += (size_t)NN * UU;
    float* agg  = ws + off; off += (size_t)NN * UU;
    float* tn   = ws + off; off += (size_t)NN * UU;
    float* wbuf = ws + off; off += (size_t)EE * UU;
    float* w1   = ws + off; off += (size_t)EE * UU;
    float* stats = ws + off; off += 256;   // [sumN,sumsqN,sumE,sumsqE,hsum,...]
    float* sumN   = stats;
    float* sumsqN = stats + 32;
    float* sumE   = stats + 64;
    float* sumsqE = stats + 96;
    float* hsum   = stats + 128;
    float* y1 = ws + off; off += (size_t)64  * NN;          // [64,20000]
    float* o1 = ws + off; off += (size_t)64  * (NN / 2);    // [64,10000]
    float* y2 = ws + off; off += (size_t)256 * (NN / 2);    // [256,10000]
    float* o2 = ws + off; off += (size_t)256 * (NN / 4);    // [256,5000]
    float* y3 = ws + off; off += (size_t)128 * (NN / 4);    // [128,5000]
    float* cst = ws + off; off += 1024;  // conv stats
    float* m1 = cst, *v1 = cst + 64, *m2 = cst + 128, *v2 = cst + 384, *m3 = cst + 640, *v3 = cst + 768;
    float* o3mean = ws + off; off += 128;

    const int TB = 256;

    // ---- degree ----
    k_zero<<<ceil_div(NN, TB), TB, 0, stream>>>(deg, NN);
    k_degree<<<ceil_div(EE, TB), TB, 0, stream>>>(src, deg, EE);

    // ---- embeddings ----
    k_node_embed<<<ceil_div(NN * UU, TB), TB, 0, stream>>>(x, v_lin0_w, v_lin0_b, h, NN);
    k_edge_embed<<<ceil_div(EE * UU, TB), TB, 0, stream>>>(edge_attr, e_lin0_w, e_lin0_b, wbuf, EE);

    // ---- GNN layers ----
    const int node_gemm_blocks = ceil_div(NN / 16, 8);   // 8 waves / 256-thread block
    const int edge_gemm_blocks = ceil_div(EE / 16, 8);
    for (int i = 0; i < DD; ++i) {
        k_zero<<<ceil_div(NN * UU, TB), TB, 0, stream>>>(agg, NN * UU);
        k_zero<<<1, 128, 0, stream>>>(stats, 128);

        k_node_gemm4<<<node_gemm_blocks, TB, 0, stream>>>(
            h, vW + (size_t)i * 4 * UU * UU, vB + (size_t)i * 4 * UU, x1, x2, x3, x4, NN);

        k_edge_gemm_msg<<<edge_gemm_blocks, TB, 0, stream>>>(
            wbuf, eW + (size_t)i * UU * UU, eB + (size_t)i * UU, w1, src, dst, x2, agg, EE);

        k_node_prebn<<<512, TB, 0, stream>>>(x1, agg, deg, tn, sumN, sumsqN, NN);
        k_bn_silu_residual<<<ceil_div(NN * UU, TB), TB, 0, stream>>>(
            h, tn, sumN, sumsqN, v_bn_g + i * UU, v_bn_b + i * UU, NN);

        k_edge_prebn<<<2048, TB, 0, stream>>>(w1, x3, x4, src, dst, sumE, sumsqE, EE);
        k_bn_silu_residual<<<ceil_div(EE * UU, TB), TB, 0, stream>>>(
            wbuf, w1, sumE, sumsqE, e_bn_g + i * UU, e_bn_b + i * UU, EE);
    }

    // ---- prob_embed = h.mean(0): accumulate column sums ----
    k_zero<<<1, 64, 0, stream>>>(hsum, 64);
    k_colsum<<<512, TB, 0, stream>>>(h, NN, hsum);

    // ---- CNN branch ----
    k_conv1d3<<<ceil_div(64 * NN, TB), TB, 0, stream>>>(pop, conv1_w, conv1_b, y1, PP, NN, 64);
    k_stats_rows<<<64, TB, 0, stream>>>(y1, NN, m1, v1);
    k_bn_silu_pool<<<ceil_div(64 * (NN / 2), TB), TB, 0, stream>>>(y1, m1, v1, bn1_g, bn1_b, o1, 64, NN);

    k_conv1d3<<<ceil_div(256 * (NN / 2), TB), TB, 0, stream>>>(o1, conv2_w, conv2_b, y2, 64, NN / 2, 256);
    k_stats_rows<<<256, TB, 0, stream>>>(y2, NN / 2, m2, v2);
    k_bn_silu_pool<<<ceil_div(256 * (NN / 4), TB), TB, 0, stream>>>(y2, m2, v2, bn2_g, bn2_b, o2, 256, NN / 2);

    k_conv1d3<<<ceil_div(128 * (NN / 4), TB), TB, 0, stream>>>(o2, conv3_w, conv3_b, y3, 256, NN / 4, 128);
    k_stats_rows<<<128, TB, 0, stream>>>(y3, NN / 4, m3, v3);
    k_bn_silu_mean<<<128, TB, 0, stream>>>(y3, m3, v3, bn3_g, bn3_b, o3mean, NN / 4);

    // ---- combined head ----
    k_head<<<1, PP, 0, stream>>>(o3mean, fc_w, fc_b, hsum, lin1_w, lin1_b,
                                 cbn_g, cbn_b, lin2_w, lin2_b, (float*)d_out);
}
